// SelfAttention_73761768341899
// MI455X (gfx1250) — matrix operand order repaired and checked by reading, not compiled, not run on hardware
//
#include <hip/hip_runtime.h>

// ---------------------------------------------------------------------------
// CDNA5 (gfx1250) wave32 WMMA self-attention
//   D_MODEL=1024, N_HEADS=16, HEAD_DIM=64, B=2, L=2048 (NT=16 * LD=128)
// All matmuls use v_wmma_f32_16x16x32_bf16 (bf16 in, f32 accumulate).
// Fragment loads are arranged as contiguous 16B runs -> ds_load_b128 /
// global_load_b128 instead of scalar u16 gathers.
// ---------------------------------------------------------------------------

typedef __attribute__((ext_vector_type(16))) __bf16 v16bf;
typedef __attribute__((ext_vector_type(8)))  float  v8f;
typedef __bf16 bf16_t;

#define WMMA_BF16(A, B, C) \
  __builtin_amdgcn_wmma_f32_16x16x32_bf16(false, (A), false, (B), (short)0, (C), false, false)

static __device__ __forceinline__ bf16_t f2bf(float f) {
  union { float f; unsigned int u; } in;
  in.f = f;
  unsigned int u = in.u;
  u += 0x7FFFu + ((u >> 16) & 1u);   // round-to-nearest-even
  union { unsigned short u; bf16_t b; } out;
  out.u = (unsigned short)(u >> 16);
  return out.b;
}

static __device__ __forceinline__ v8f v8f_zero() {
  v8f z;
#pragma unroll
  for (int i = 0; i < 8; ++i) z[i] = 0.0f;
  return z;
}

// Load a 16-element bf16 fragment as two 16-byte runs (p0 -> elems 0..7,
// p1 -> elems 8..15).  Works for LDS and global pointers.
union FragU { uint4 u[2]; v16bf v; };
static __device__ __forceinline__ v16bf load_frag(const bf16_t* p0, const bf16_t* p1) {
  FragU f;
  f.u[0] = *(const uint4*)p0;
  f.u[1] = *(const uint4*)p1;
  return f.v;
}

// Convert float4 -> 4 bf16 and store as a single 8-byte LDS store.
static __device__ __forceinline__ void store4bf(bf16_t* p, float4 f) {
  union { bf16_t b[4]; uint2 u; } t;
  t.b[0] = f2bf(f.x); t.b[1] = f2bf(f.y); t.b[2] = f2bf(f.z); t.b[3] = f2bf(f.w);
  *(uint2*)p = t.u;
}

#define D_MODEL 1024
#define SEQ_L   2048
#define APITCH  40   // bf16 pitch (80 B) keeps all 16B runs aligned

// ---------------------------------------------------------------------------
// Kernel 1: fused QKV projection.  y = x @ W^T + b for W in {Wq,Wk,Wv}.
// Block tile 128(M) x 128(N), 8 waves (4x2), wave tile 32x64 (2x4 WMMA), BK=32.
// grid: (3072/128, 4096/128) = (24, 32); 256 threads.
// ---------------------------------------------------------------------------
__global__ __launch_bounds__(256) void gemm_qkv_kernel(
    const float* __restrict__ x,
    const float* __restrict__ Wq, const float* __restrict__ bq,
    const float* __restrict__ Wk, const float* __restrict__ bk,
    const float* __restrict__ Wv, const float* __restrict__ bv,
    float* __restrict__ qraw, float* __restrict__ kraw,
    float* __restrict__ v_out, bf16_t* __restrict__ vbf) {
  __shared__ __align__(16) bf16_t As[128][APITCH];  // [M][K]
  __shared__ __align__(16) bf16_t Bs[128][APITCH];  // [N][K] == W's natural layout

  const int tid  = threadIdx.x;
  const int lane = tid & 31;
  const int wave = tid >> 5;
  const int wm   = wave >> 1;      // 0..3  (32 rows each)
  const int wn   = wave & 1;       // 0..1  (64 cols each)
  const int h16  = lane >> 4;
  const int ln   = lane & 15;

  const int mbase = blockIdx.y * 128;
  const int nbase = blockIdx.x * 128;
  const int which = nbase >> 10;           // 0=q 1=k 2=v
  const int ncol  = nbase & 1023;
  const float* W    = (which == 0) ? Wq : (which == 1) ? Wk : Wv;
  const float* bias = (which == 0) ? bq : (which == 1) ? bk : bv;

  v8f acc[2][4];
#pragma unroll
  for (int i = 0; i < 2; ++i)
#pragma unroll
    for (int j = 0; j < 4; ++j) acc[i][j] = v8f_zero();

  for (int kk = 0; kk < D_MODEL; kk += 32) {
    // stage A (128x32) and B (128x32) fp32 -> bf16 (float4 loads, b64 stores)
#pragma unroll
    for (int it = 0; it < 4; ++it) {
      int idx = it * 256 + tid;            // 0..1023
      int row = idx >> 3, c4 = (idx & 7) * 4;
      float4 fa = *(const float4*)(x + (size_t)(mbase + row) * D_MODEL + kk + c4);
      store4bf(&As[row][c4], fa);
      float4 fb = *(const float4*)(W + (size_t)(ncol + row) * D_MODEL + kk + c4);
      store4bf(&Bs[row][c4], fb);
    }
    __syncthreads();

    // A frags: two contiguous 8-elem runs per lane
    const bf16_t* ar0 = &As[wm * 32 + ln][0];
    const bf16_t* ar1 = &As[wm * 32 + 16 + ln][0];
    v16bf a0 = load_frag(ar0 + 16 * h16, ar0 + 16 * h16 + 8);
    v16bf a1 = load_frag(ar1 + 16 * h16, ar1 + 16 * h16 + 8);
    // B frags: 16 contiguous elems starting at K = 16*h16
#pragma unroll
    for (int t = 0; t < 4; ++t) {
      const bf16_t* br = &Bs[wn * 64 + t * 16 + ln][16 * h16];
      v16bf b = load_frag(br, br + 8);
      acc[0][t] = WMMA_BF16(a0, b, acc[0][t]);
      acc[1][t] = WMMA_BF16(a1, b, acc[1][t]);
    }
    __syncthreads();
  }

  // epilogue
#pragma unroll
  for (int tm = 0; tm < 2; ++tm) {
#pragma unroll
    for (int t = 0; t < 4; ++t) {
#pragma unroll
      for (int j = 0; j < 8; ++j) {
        int m  = mbase + wm * 32 + tm * 16 + j + 8 * h16;
        int nb = ncol + wn * 64 + t * 16 + ln;
        float val = acc[tm][t][j] + bias[nb];
        if (which == 0) {
          qraw[(size_t)m * D_MODEL + nb] = val;
        } else if (which == 1) {
          kraw[(size_t)m * D_MODEL + nb] = val;
        } else {
          int b_ = m >> 11, l = m & 2047;
          int hh = nb >> 6, hd = nb & 63;
          size_t o = ((size_t)(b_ * 16 + hh) * SEQ_L + l) * 64 + hd;
          v_out[o] = val;
          vbf[o]   = f2bf(val);
        }
      }
    }
  }
}

// ---------------------------------------------------------------------------
// Kernel 2: RoPE on q and k -> bf16 (B,H,L,HD) for attention + fp32 roped k.
// grid: 4194304/256 = 16384
// ---------------------------------------------------------------------------
__global__ __launch_bounds__(256) void rope_kernel(
    const float* __restrict__ qraw, const float* __restrict__ kraw,
    const int* __restrict__ dt,
    bf16_t* __restrict__ qb, bf16_t* __restrict__ kb,
    float* __restrict__ k_out) {
  int idx = blockIdx.x * 256 + threadIdx.x;   // [0, 4096*1024)
  int d = idx & 1023;
  int m = idx >> 10;
  int hh = d >> 6, hd = d & 63;
  int b_ = m >> 11, l = m & 2047;
  int j = hd & 31;

  float t   = (float)dt[l >> 7];
  float inv = __expf(-(float)j * (11.512925464970229f / 32.0f));  // base^(-j/32)
  float fr  = t * inv;
  float c = cosf(fr), s = sinf(fr);

  size_t base = (size_t)m * D_MODEL + hh * 64;
  int part = (hd < 32) ? hd + 32 : hd - 32;

  float qv = qraw[idx];
  float qp = qraw[base + part];
  float qr = (hd < 32) ? -qp : qp;
  float q_ro = qv * c + qr * s;

  float kv = kraw[idx];
  float kp = kraw[base + part];
  float kr = (hd < 32) ? -kp : kp;
  float k_ro = kv * c + kr * s;

  size_t o = ((size_t)(b_ * 16 + hh) * SEQ_L + l) * 64 + hd;
  qb[o]    = f2bf(q_ro);
  kb[o]    = f2bf(k_ro);
  k_out[o] = k_ro;
}

// ---------------------------------------------------------------------------
// Kernel 3: flash attention.  One wave = 16 query rows; block = 8 waves over
// one (b,h); 2048 keys in 32-key LDS chunks.  V staged transposed so PV
// B-frags are contiguous.  8 WMMA per chunk per wave.
// grid: B*H*(L/128) = 512; 256 threads.
// ---------------------------------------------------------------------------
__global__ __launch_bounds__(256) void attn_kernel(
    const bf16_t* __restrict__ qb, const bf16_t* __restrict__ kbm,
    const bf16_t* __restrict__ vbm, bf16_t* __restrict__ aob) {
  __shared__ __align__(16) bf16_t Ks[32][64];          // [key][hd]
  __shared__ __align__(16) bf16_t Vst[64][APITCH];     // [hd][key]  (transposed)
  __shared__ __align__(16) bf16_t Ps[8][16][APITCH];   // per-wave P tile [m][key]

  const int tid  = threadIdx.x;
  const int lane = tid & 31;
  const int w    = tid >> 5;
  const int h16  = lane >> 4;
  const int ln   = lane & 15;

  const int bh    = blockIdx.x >> 4;   // b*16+h
  const int qblk  = blockIdx.x & 15;
  const int b_    = bh >> 4;
  const int hh    = bh & 15;
  const size_t seq_base = (size_t)bh * SEQ_L * 64;
  const int qbase = qblk * 128 + w * 16;

  // Q fragments (A layout): two 16B runs per lane, loaded once from global
  const bf16_t* qrow = qb + seq_base + (size_t)(qbase + ln) * 64;
  v16bf qa0 = load_frag(qrow + 16 * h16, qrow + 16 * h16 + 8);
  v16bf qa1 = load_frag(qrow + 32 + 16 * h16, qrow + 40 + 16 * h16);

  float mrun[8], lrun[8];
#pragma unroll
  for (int j = 0; j < 8; ++j) { mrun[j] = -1e30f; lrun[j] = 0.0f; }
  v8f o0 = v8f_zero(), o1 = v8f_zero(), o2 = v8f_zero(), o3 = v8f_zero();

  const int key = tid >> 3;            // staging: 32 keys x 8 hd-chunks
  const int c8  = (tid & 7) * 8;

  for (int kc = 0; kc < 64; ++kc) {
    const int key0 = kc * 32;
    {
      const bf16_t* kg = kbm + seq_base + (size_t)(key0 + key) * 64 + c8;
      const bf16_t* vg = vbm + seq_base + (size_t)(key0 + key) * 64 + c8;
      *(uint4*)(&Ks[key][c8]) = *(const uint4*)kg;      // K: direct copy
      union { uint4 u; bf16_t b[8]; } tv;               // V: transpose into LDS
      tv.u = *(const uint4*)vg;
#pragma unroll
      for (int e = 0; e < 8; ++e) Vst[c8 + e][key] = tv.b[e];
      if (kc < 63) {                                    // prefetch next chunk
        __builtin_prefetch(kg + 32 * 64, 0, 3);
        __builtin_prefetch(vg + 32 * 64, 0, 3);
      }
    }
    __syncthreads();

    // S = Q K^T (two 16x16 key tiles; HD=64 = two K=32 steps)
    v8f s0 = v8f_zero(), s1 = v8f_zero();
    {
      const bf16_t* kr0 = &Ks[ln][16 * h16];
      const bf16_t* kr1 = &Ks[16 + ln][16 * h16];
      v16bf f;
      f  = load_frag(kr0, kr0 + 8);       s0 = WMMA_BF16(qa0, f, s0);
      f  = load_frag(kr0 + 32, kr0 + 40); s0 = WMMA_BF16(qa1, f, s0);
      f  = load_frag(kr1, kr1 + 8);       s1 = WMMA_BF16(qa0, f, s1);
      f  = load_frag(kr1 + 32, kr1 + 40); s1 = WMMA_BF16(qa1, f, s1);
    }

    // online softmax (row = j + 8*h16; columns across 16-lane groups)
    float alpha[8];
#pragma unroll
    for (int j = 0; j < 8; ++j) {
      float x0 = s0[j] * 0.125f;           // 1/sqrt(HEAD_DIM)
      float x1 = s1[j] * 0.125f;
      float mx = fmaxf(x0, x1);
      mx = fmaxf(mx, __shfl_xor(mx, 1, 32));
      mx = fmaxf(mx, __shfl_xor(mx, 2, 32));
      mx = fmaxf(mx, __shfl_xor(mx, 4, 32));
      mx = fmaxf(mx, __shfl_xor(mx, 8, 32));
      float mnew = fmaxf(mrun[j], mx);
      float al = __expf(mrun[j] - mnew);
      float p0 = __expf(x0 - mnew);
      float p1 = __expf(x1 - mnew);
      float rs = p0 + p1;
      rs += __shfl_xor(rs, 1, 32);
      rs += __shfl_xor(rs, 2, 32);
      rs += __shfl_xor(rs, 4, 32);
      rs += __shfl_xor(rs, 8, 32);
      lrun[j]  = lrun[j] * al + rs;
      mrun[j]  = mnew;
      alpha[j] = al;
      Ps[w][j + 8 * h16][ln]      = f2bf(p0);
      Ps[w][j + 8 * h16][16 + ln] = f2bf(p1);
    }
#pragma unroll
    for (int j = 0; j < 8; ++j) {
      o0[j] *= alpha[j]; o1[j] *= alpha[j]; o2[j] *= alpha[j]; o3[j] *= alpha[j];
    }

    // O += P V : A-frag from Ps (two 16B runs), B-frags contiguous from Vst
    const bf16_t* pp = &Ps[w][ln][0];
    v16bf pa = load_frag(pp + 16 * h16, pp + 16 * h16 + 8);
    {
      const bf16_t* v0 = &Vst[ln][16 * h16];
      const bf16_t* v1 = &Vst[16 + ln][16 * h16];
      const bf16_t* v2 = &Vst[32 + ln][16 * h16];
      const bf16_t* v3 = &Vst[48 + ln][16 * h16];
      o0 = WMMA_BF16(pa, load_frag(v0, v0 + 8), o0);
      o1 = WMMA_BF16(pa, load_frag(v1, v1 + 8), o1);
      o2 = WMMA_BF16(pa, load_frag(v2, v2 + 8), o2);
      o3 = WMMA_BF16(pa, load_frag(v3, v3 + 8), o3);
    }
    __syncthreads();
  }

  // normalize and store attention output in flat (M, D) bf16 for the O-proj
#pragma unroll
  for (int j = 0; j < 8; ++j) {
    float inv = 1.0f / lrun[j];
    int qrow_g = qbase + j + 8 * h16;
    size_t mflat = ((size_t)(b_ * SEQ_L + qrow_g)) * D_MODEL + hh * 64;
    aob[mflat + ln]      = f2bf(o0[j] * inv);
    aob[mflat + 16 + ln] = f2bf(o1[j] * inv);
    aob[mflat + 32 + ln] = f2bf(o2[j] * inv);
    aob[mflat + 48 + ln] = f2bf(o3[j] * inv);
  }
}

// ---------------------------------------------------------------------------
// Kernel 4: output projection.  out = attn @ Wo^T + bo (A already bf16).
// Block tile 128x128, wave tile 32x64.  grid: (8, 32); 256 threads.
// ---------------------------------------------------------------------------
__global__ __launch_bounds__(256) void gemm_out_kernel(
    const bf16_t* __restrict__ aob, const float* __restrict__ Wo,
    const float* __restrict__ bo, float* __restrict__ out) {
  __shared__ __align__(16) bf16_t As[128][APITCH];
  __shared__ __align__(16) bf16_t Bs[128][APITCH];

  const int tid  = threadIdx.x;
  const int lane = tid & 31;
  const int wave = tid >> 5;
  const int wm = wave >> 1, wn = wave & 1;
  const int h16 = lane >> 4, ln = lane & 15;

  const int mbase = blockIdx.y * 128;
  const int nbase = blockIdx.x * 128;

  v8f acc[2][4];
#pragma unroll
  for (int i = 0; i < 2; ++i)
#pragma unroll
    for (int j = 0; j < 4; ++j) acc[i][j] = v8f_zero();

  for (int kk = 0; kk < D_MODEL; kk += 32) {
    // A already bf16: straight 16B copies
#pragma unroll
    for (int it = 0; it < 2; ++it) {
      int idx = it * 256 + tid;            // 0..511
      int row = idx >> 2, c8 = (idx & 3) * 8;
      *(uint4*)(&As[row][c8]) =
          *(const uint4*)(aob + (size_t)(mbase + row) * D_MODEL + kk + c8);
    }
#pragma unroll
    for (int it = 0; it < 4; ++it) {
      int idx = it * 256 + tid;            // 0..1023
      int n = idx >> 3, c4 = (idx & 7) * 4;
      float4 fb = *(const float4*)(Wo + (size_t)(nbase + n) * D_MODEL + kk + c4);
      store4bf(&Bs[n][c4], fb);
    }
    __syncthreads();

    const bf16_t* ar0 = &As[wm * 32 + ln][0];
    const bf16_t* ar1 = &As[wm * 32 + 16 + ln][0];
    v16bf a0 = load_frag(ar0 + 16 * h16, ar0 + 16 * h16 + 8);
    v16bf a1 = load_frag(ar1 + 16 * h16, ar1 + 16 * h16 + 8);
#pragma unroll
    for (int t = 0; t < 4; ++t) {
      const bf16_t* br = &Bs[wn * 64 + t * 16 + ln][16 * h16];
      v16bf b = load_frag(br, br + 8);
      acc[0][t] = WMMA_BF16(a0, b, acc[0][t]);
      acc[1][t] = WMMA_BF16(a1, b, acc[1][t]);
    }
    __syncthreads();
  }

#pragma unroll
  for (int tm = 0; tm < 2; ++tm) {
#pragma unroll
    for (int t = 0; t < 4; ++t) {
#pragma unroll
      for (int j = 0; j < 8; ++j) {
        int m  = mbase + wm * 32 + tm * 16 + j + 8 * h16;
        int nb = nbase + wn * 64 + t * 16 + ln;
        out[(size_t)m * D_MODEL + nb] = acc[tm][t][j] + bo[nb];
      }
    }
  }
}

// ---------------------------------------------------------------------------
extern "C" void kernel_launch(void* const* d_in, const int* in_sizes, int n_in,
                              void* d_out, int out_size, void* d_ws, size_t ws_size,
                              hipStream_t stream) {
  (void)in_sizes; (void)n_in; (void)out_size; (void)ws_size;

  const float* x  = (const float*)d_in[0];
  const int*   dt = (const int*)d_in[1];
  const float* Wq = (const float*)d_in[2];
  const float* bq = (const float*)d_in[3];
  const float* Wk = (const float*)d_in[4];
  const float* bk = (const float*)d_in[5];
  const float* Wv = (const float*)d_in[6];
  const float* bv = (const float*)d_in[7];
  const float* Wo = (const float*)d_in[8];
  const float* bo = (const float*)d_in[9];

  float* out   = (float*)d_out;             // (B,NT,LD,D)  = 4,194,304 f32
  float* k_out = out + 4194304;             // (B,H,L,HD)
  float* v_out = out + 8388608;             // (B,H,L,HD)

  char* ws = (char*)d_ws;                   // 64 MB used
  float*  qraw = (float*)(ws);                          // 16 MB
  float*  kraw = (float*)(ws + (size_t)16 * 1048576);   // 16 MB
  bf16_t* qb   = (bf16_t*)(ws + (size_t)32 * 1048576);  //  8 MB (B,H,L,HD)
  bf16_t* kb   = (bf16_t*)(ws + (size_t)40 * 1048576);  //  8 MB
  bf16_t* vb   = (bf16_t*)(ws + (size_t)48 * 1048576);  //  8 MB
  bf16_t* aob  = (bf16_t*)(ws + (size_t)56 * 1048576);  //  8 MB (M,D)

  gemm_qkv_kernel<<<dim3(24, 32), 256, 0, stream>>>(
      x, Wq, bq, Wk, bk, Wv, bv, qraw, kraw, v_out, vb);
  rope_kernel<<<16384, 256, 0, stream>>>(qraw, kraw, dt, qb, kb, k_out);
  attn_kernel<<<512, 256, 0, stream>>>(qb, kb, vb, aob);
  gemm_out_kernel<<<dim3(8, 32), 256, 0, stream>>>(aob, Wo, bo, out);
}
